// GRUCDE_50311246905431
// MI455X (gfx1250) — matrix operand order repaired and checked by reading, not compiled, not run on hardware
//
#include <hip/hip_runtime.h>
#include <stdint.h>

// ---------------------------------------------------------------------------
// GRU-ODE (neural CDE) for MI455X / gfx1250.
// 4 persistent workgroups x 512 threads (16 wave32's). Each workgroup owns 16
// batch columns -> every mat-vec becomes a 16-column GEMM executed with
// v_wmma_f32_16x16x32_bf16 (f32 accumulate). Weights pre-packed into the
// CDNA5 A-matrix VGPR striping; activations packed to the B layout in LDS so
// WMMA operand feeds are ds_load_b128 (static __shared__ keeps addrspace 3).
// ---------------------------------------------------------------------------

typedef __attribute__((ext_vector_type(16))) __bf16 v16bf;
typedef __attribute__((ext_vector_type(8)))  float  v8f;

#define B_    64
#define T_    200
#define LIN   100
#define S_    8
#define XC_   16
#define H_    256
#define W_    128
#define O_    8
#define NINT  99      // number of ODE intervals (len(dts) = T-L-1)
#define G3    768     // 3*H
#define BQ    16      // batch columns per workgroup
#define NTH   512     // 16 waves

#if __has_builtin(__builtin_amdgcn_tanhf)
  #define TANHF_(x) __builtin_amdgcn_tanhf(x)
#elif __has_builtin(__builtin_amdgcn_tanh_f32)
  #define TANHF_(x) __builtin_amdgcn_tanh_f32(x)
#else
  #define TANHF_(x) tanhf(x)
#endif

// bf16 round-to-nearest-even
__device__ __forceinline__ unsigned short f32_to_bf16(float f) {
  unsigned int u = __float_as_uint(f);
  u += 0x7fffu + ((u >> 16) & 1u);
  return (unsigned short)(u >> 16);
}

// Write one element of a K x 16 B-matrix into packed WMMA B layout.
// Per 32-K chunk (512 bf16): lane n (n<16) elems 0..7 = K 0..7, 8..15 = K 16..23
// of column n; lane n+16 elems 0..7 = K 8..15, 8..15 = K 24..31 of column n.
__device__ __forceinline__ void pack_b(unsigned short* bpk, int kk, int col, float v) {
  int chunk = kk >> 5;
  int r     = kk & 31;
  int lane  = ((r >> 3) & 1) ? (col + 16) : col;
  bpk[(chunk * 32 + lane) * 16 + ((r & 7) | ((r >> 4) << 3))] = f32_to_bf16(v);
}

__device__ __forceinline__ v16bf ld16(const unsigned short* p, int lane) {
  return ((const v16bf*)p)[lane];
}

__device__ __forceinline__ v8f wmma_bf16(v16bf a, v16bf b, v8f c) {
  return __builtin_amdgcn_wmma_f32_16x16x32_bf16(false, a, false, b, (short)0, c,
                                                 false, false);
}

__device__ __forceinline__ float sigmoidf_(float x) { return 1.0f / (1.0f + __expf(-x)); }
__device__ __forceinline__ float softplusf_(float x) {
  return (x > 20.0f) ? x : __logf(1.0f + __expf(x));
}

// dopri5 Butcher tableau (stage coefficients)
static __device__ constexpr float DOPRI_A[6][5] = {
  {0.f, 0.f, 0.f, 0.f, 0.f},
  {0.2f, 0.f, 0.f, 0.f, 0.f},
  {3.f/40.f, 9.f/40.f, 0.f, 0.f, 0.f},
  {44.f/45.f, -56.f/15.f, 32.f/9.f, 0.f, 0.f},
  {19372.f/6561.f, -25360.f/2187.f, 64448.f/6561.f, -212.f/729.f, 0.f},
  {9017.f/3168.f, -355.f/33.f, 46732.f/5247.f, 49.f/176.f, -5103.f/18656.f}
};

// ---------------------------------------------------------------------------
// Pre-pack fp32 row-major (rows x K) weights into WMMA A-matrix layout, bf16.
// Tile = 16 rows, chunk = 32 K. Per chunk 512 bf16: lane L<16 holds row L,
// elems 0..7 = K 0..7, elems 8..15 = K 16..23; lane L+16 holds row L,
// elems 0..7 = K 8..15, elems 8..15 = K 24..31. (K beyond K_real zero-padded.)
// ---------------------------------------------------------------------------
__global__ void repack_a_kernel(const float* __restrict__ src,
                                unsigned short* __restrict__ dst,
                                int tiles, int nchunks, int K_src, int K_real,
                                int rows) {
  int id = blockIdx.x * blockDim.x + threadIdx.x;
  int total = tiles * nchunks * 512;
  if (id >= total) return;
  int elem = id & 15;
  int lane = (id >> 4) & 31;
  int tc   = id >> 9;
  int chunkid = tc % nchunks;
  int tile    = tc / nchunks;
  int row = tile * 16 + (lane & 15);
  int kin = (lane < 16) ? ((elem < 8) ? elem : elem + 8)
                        : ((elem < 8) ? elem + 8 : elem + 16);
  int kk = chunkid * 32 + kin;
  float v = (row < rows && kk < K_real) ? src[row * K_src + kk] : 0.0f;
  dst[id] = f32_to_bf16(v);
}

// ---------------------------------------------------------------------------
// Main persistent kernel: GRU encoder (100 steps) + dopri5 decoder (99
// intervals x 2 substeps x 6 stages), all WMMA bf16 with f32 accumulate.
// Static shared total ~146KB (encoder scratch unioned with W0 tile store).
// ---------------------------------------------------------------------------
__global__ __launch_bounds__(NTH)
void grucde_main(const float* __restrict__ y_past, const float* __restrict__ t_arr,
                 const float* __restrict__ cx,
                 const float* __restrict__ gru_b, const float* __restrict__ gru_bn,
                 const float* __restrict__ mb0, const float* __restrict__ mb1,
                 const float* __restrict__ mb2,
                 const float* __restrict__ ro_w, const float* __restrict__ ro_b,
                 const unsigned short* __restrict__ wih_pk,
                 const unsigned short* __restrict__ whh_pk,
                 const unsigned short* __restrict__ w0_pk,
                 const unsigned short* __restrict__ w1_pk,
                 const unsigned short* __restrict__ w2_pk,
                 float* __restrict__ out)
{
  // Encoder scratch (~61KB) and decoder W0 tile store (64KB) have disjoint
  // live ranges -> union them to stay well under the per-WG LDS budget.
  __shared__ union {
    struct {
      float h[H_ * BQ];                       // GRU hidden (f32)
      float r[H_ * BQ];                       // r gate
      float u[H_ * BQ];                       // u gate
      alignas(32) unsigned short hpk[8 * 512]; // h packed as B (K=256)
      alignas(32) unsigned short xpk[512];     // x_t packed as B (K=32)
      float gb[G3];
      float gbn[H_];
    } enc;
    alignas(32) unsigned short w0[8 * 8 * 512]; // W0 packed tiles (64KB)
  } U;
  __shared__ alignas(32) unsigned short s_w1[8 * 4 * 512];   // 32KB
  __shared__ float s_b2[H_ * XC_];                            // 16KB
  __shared__ float s_b0[W_];
  __shared__ float s_b1[W_];
  __shared__ alignas(32) unsigned short s_zspk[8 * 512];      // stage in (K=256)
  __shared__ alignas(32) unsigned short s_a0pk[4 * 512];      // K=128
  __shared__ alignas(32) unsigned short s_a1pk[4 * 512];      // K=128
  __shared__ float s_kz[H_ * BQ];   // k during RK stages / z at readout (16KB)
  __shared__ float s_dx[XC_ * BQ];  // dxdt 16x16

  const int tid  = threadIdx.x;
  const int wave = tid >> 5;
  const int lane = tid & 31;
  const int bb   = blockIdx.x * BQ;   // first batch column of this workgroup
  const int col  = lane & 15;
  const int roff = (lane >> 4) * 8;

  // ---- init ----
  if (tid < G3) U.enc.gb[tid]  = gru_b[tid];
  if (tid < H_) U.enc.gbn[tid] = gru_bn[tid];
  for (int i = tid; i < H_ * BQ; i += NTH) U.enc.h[i] = 0.0f;
  for (int i = tid; i < 8 * 512; i += NTH) U.enc.hpk[i] = 0;
  __syncthreads();

  // =================== GRU encoder: 100 sequential steps ===================
  for (int step = 0; step < LIN; ++step) {
    // build x_t = [y_past(8) | coeffs_x(15) | t(1)] padded to K=32, pack as B
    if (tid < 24 * BQ) {
      int c = tid / 24, idx = tid - c * 24;
      int b = bb + c;
      float v;
      if (idx < S_)      v = y_past[(b * LIN + step) * S_ + idx];
      else if (idx < 23) v = cx[(b * T_ + step) * (XC_ - 1) + (idx - S_)];
      else               v = t_arr[step];
      pack_b(U.enc.xpk, idx, c, v);
    } else if (tid < 24 * BQ + 8 * BQ) {
      int q = tid - 24 * BQ;
      pack_b(U.enc.xpk, 24 + (q & 7), q >> 3, 0.0f);
    }
    __syncthreads();

    const v16bf bx = ld16(U.enc.xpk, lane);

    // pass 1: r/z gate rows (tiles 0..31): ig+hg fused in one accumulator
    for (int tt = wave; tt < 32; tt += 16) {
      v8f acc = {};
      acc = wmma_bf16(ld16(wih_pk + tt * 512, lane), bx, acc);
      #pragma unroll
      for (int cc = 0; cc < 8; ++cc)
        acc = wmma_bf16(ld16(whh_pk + (tt * 8 + cc) * 512, lane),
                        ld16(U.enc.hpk + cc * 512, lane), acc);
      #pragma unroll
      for (int j = 0; j < 8; ++j) {
        int row = tt * 16 + roff + j;
        float g = sigmoidf_(acc[j] + U.enc.gb[row]);
        if (row < H_) U.enc.r[row * BQ + col]        = g;
        else          U.enc.u[(row - H_) * BQ + col] = g;
      }
    }
    __syncthreads();

    // pass 2: n gate rows (tiles 32..47), input/hidden parts kept separate
    {
      int tt = 32 + wave;
      v8f acc_i = {}, acc_h = {};
      acc_i = wmma_bf16(ld16(wih_pk + tt * 512, lane), bx, acc_i);
      #pragma unroll
      for (int cc = 0; cc < 8; ++cc)
        acc_h = wmma_bf16(ld16(whh_pk + (tt * 8 + cc) * 512, lane),
                          ld16(U.enc.hpk + cc * 512, lane), acc_h);
      float hnew[8];
      #pragma unroll
      for (int j = 0; j < 8; ++j) {
        int row = tt * 16 + roff + j;
        int hh  = row - 2 * H_;
        float ia = acc_i[j] + U.enc.gb[row];
        float hn = acc_h[j] + U.enc.gbn[hh];
        float r  = U.enc.r[hh * BQ + col];
        float n  = TANHF_(ia + r * hn);
        float u  = U.enc.u[hh * BQ + col];
        hnew[j]  = n + u * (U.enc.h[hh * BQ + col] - n);
      }
      __syncthreads();   // all waves done reading hpk/h before overwrite
      #pragma unroll
      for (int j = 0; j < 8; ++j) {
        int hh = wave * 16 + roff + j;
        U.enc.h[hh * BQ + col] = hnew[j];
        pack_b(U.enc.hpk, hh, col, hnew[j]);
      }
    }
    __syncthreads();
  }

  // ---- z0 = h: readout index 0, z into per-thread registers ----
  float zreg[8];
  #pragma unroll
  for (int i = 0; i < 8; ++i) zreg[i] = U.enc.h[tid * 8 + i];
  if (tid < BQ * O_) {
    int c = tid >> 3, o = tid & 7;
    float acc = ro_b[o];
    for (int hh = 0; hh < H_; ++hh) acc += ro_w[o * H_ + hh] * U.enc.h[hh * BQ + c];
    out[((bb + c) * (NINT + 1)) * O_ + o] = acc;
  }
  __syncthreads();   // encoder region dead from here; reuse as W0 store

  // MLP weights/biases -> LDS resident
  for (int i = tid; i < 8 * 8 * 512; i += NTH) U.w0[i] = w0_pk[i];
  for (int i = tid; i < 8 * 4 * 512; i += NTH) s_w1[i] = w1_pk[i];
  for (int i = tid; i < H_ * XC_;    i += NTH) s_b2[i] = mb2[i];
  if (tid < W_) { s_b0[tid] = mb0[tid]; s_b1[tid] = mb1[tid]; }
  __syncthreads();

  float kreg[6][8];
  #pragma unroll
  for (int s = 0; s < 6; ++s)
    #pragma unroll
    for (int i = 0; i < 8; ++i) kreg[s][i] = 0.0f;

  // =================== Neural-CDE decoder ===================
  for (int iv = 0; iv < NINT; ++iv) {
    int tL = LIN + iv;
    float dt    = t_arr[tL + 1] - t_arr[tL];
    float hstep = 0.5f * dt;                      // N_SUB = 2
    if (tid < XC_ * BQ) {
      int c = tid >> 4, cl = tid & 15, b = bb + cl;
      float x1 = (c < 15) ? cx[(b * T_ + tL + 1) * (XC_ - 1) + c] : t_arr[tL + 1];
      float x0 = (c < 15) ? cx[(b * T_ + tL) * (XC_ - 1) + c]     : t_arr[tL];
      s_dx[c * BQ + cl] = (x1 - x0) / dt;
    }
    __syncthreads();

    for (int ss = 0; ss < 2; ++ss) {
      #pragma unroll
      for (int st = 0; st < 6; ++st) {
        // stage input zs = z + hstep * sum_j A[st][j] k_j ; pack as B (K=256)
        #pragma unroll
        for (int i = 0; i < 8; ++i) {
          int e = tid * 8 + i;
          float zs = zreg[i];
          #pragma unroll
          for (int j = 0; j < 5; ++j) zs += hstep * DOPRI_A[st][j] * kreg[j][i];
          pack_b(s_zspk, e >> 4, e & 15, zs);
        }
        __syncthreads();

        // a0 = softplus(W0 @ zs + b0)   (waves 0..7; others warm L2 for W2)
        if (wave < 8) {
          v8f acc = {};
          #pragma unroll
          for (int cc = 0; cc < 8; ++cc)
            acc = wmma_bf16(ld16(U.w0 + (wave * 8 + cc) * 512, lane),
                            ld16(s_zspk + cc * 512, lane), acc);
          #pragma unroll
          for (int j = 0; j < 8; ++j) {
            int row = wave * 16 + roff + j;
            pack_b(s_a0pk, row, col, softplusf_(acc[j] + s_b0[row]));
          }
        } else {
          __builtin_prefetch(w2_pk + (wave - 8) * 65536 + lane * 256, 0, 1);
        }
        __syncthreads();

        // a1 = softplus(W1 @ a0 + b1)
        if (wave < 8) {
          v8f acc = {};
          #pragma unroll
          for (int cc = 0; cc < 4; ++cc)
            acc = wmma_bf16(ld16(s_w1 + (wave * 4 + cc) * 512, lane),
                            ld16(s_a0pk + cc * 512, lane), acc);
          #pragma unroll
          for (int j = 0; j < 8; ++j) {
            int row = wave * 16 + roff + j;
            pack_b(s_a1pk, row, col, softplusf_(acc[j] + s_b1[row]));
          }
        }
        __syncthreads();

        // k = reshape(tanh(W2 @ a1 + b2), (H, XC)) @ dxdt
        // W2 row tile tt covers exactly h = tt, c = 0..15 -> column reduce.
        for (int q = 0; q < 16; ++q) {
          int tt = wave * 16 + q;
          v8f acc = {};
          #pragma unroll
          for (int cc = 0; cc < 4; ++cc)
            acc = wmma_bf16(ld16(w2_pk + (tt * 4 + cc) * 512, lane),
                            ld16(s_a1pk + cc * 512, lane), acc);
          float partial = 0.0f;
          #pragma unroll
          for (int j = 0; j < 8; ++j) {
            int row = tt * 16 + roff + j;
            float f = TANHF_(acc[j] + s_b2[row]);
            partial += f * s_dx[(roff + j) * BQ + col];
          }
          // combine halves (rows 0..7 live in lanes 0..15, rows 8..15 in 16..31)
          float other = __int_as_float(
              __builtin_amdgcn_ds_swizzle(__float_as_int(partial), 0x401F));
          if (lane < 16) s_kz[tt * BQ + col] = partial + other;
        }
        __syncthreads();
        #pragma unroll
        for (int i = 0; i < 8; ++i) kreg[st][i] = s_kz[tid * 8 + i];
        __syncthreads();   // required: s_kz is reused (k stages / z readout)
      }

      // z += hstep * (b1 k1 + b3 k3 + b4 k4 + b5 k5 + b6 k6)
      #pragma unroll
      for (int i = 0; i < 8; ++i)
        zreg[i] += hstep * (  (35.0f/384.0f)    * kreg[0][i]
                            + (500.0f/1113.0f)  * kreg[2][i]
                            + (125.0f/192.0f)   * kreg[3][i]
                            - (2187.0f/6784.0f) * kreg[4][i]
                            + (11.0f/84.0f)     * kreg[5][i]);
    }

    // emit y_{iv+1} = ro_w @ z + ro_b   (z staged through s_kz)
    #pragma unroll
    for (int i = 0; i < 8; ++i) s_kz[tid * 8 + i] = zreg[i];
    __syncthreads();
    if (tid < BQ * O_) {
      int c = tid >> 3, o = tid & 7;
      float acc = ro_b[o];
      for (int hh = 0; hh < H_; ++hh) acc += ro_w[o * H_ + hh] * s_kz[hh * BQ + c];
      out[((bb + c) * (NINT + 1) + (iv + 1)) * O_ + o] = acc;
    }
    __syncthreads();
  }
}

// ---------------------------------------------------------------------------
// Host side
// ---------------------------------------------------------------------------
static void launch_repack(const float* src, unsigned short* dst, int tiles,
                          int nchunks, int K, hipStream_t s) {
  int total = tiles * nchunks * 512;
  repack_a_kernel<<<(total + 255) / 256, 256, 0, s>>>(src, dst, tiles, nchunks,
                                                      K, K, tiles * 16);
}

extern "C" void kernel_launch(void* const* d_in, const int* in_sizes, int n_in,
                              void* d_out, int out_size, void* d_ws, size_t ws_size,
                              hipStream_t stream) {
  (void)in_sizes; (void)n_in; (void)out_size; (void)ws_size;
  const float* y_past = (const float*)d_in[0];
  const float* t_arr  = (const float*)d_in[1];
  const float* cxp    = (const float*)d_in[2];
  const float* wih    = (const float*)d_in[3];
  const float* whh    = (const float*)d_in[4];
  const float* gb     = (const float*)d_in[5];
  const float* gbn    = (const float*)d_in[6];
  const float* w0     = (const float*)d_in[7];
  const float* b0     = (const float*)d_in[8];
  const float* w1     = (const float*)d_in[9];
  const float* b1     = (const float*)d_in[10];
  const float* w2     = (const float*)d_in[11];
  const float* b2     = (const float*)d_in[12];
  const float* row_   = (const float*)d_in[13];
  const float* rob    = (const float*)d_in[14];
  float* out = (float*)d_out;

  // workspace layout (ushort units): packed bf16 weights, total ~1.52 MB
  unsigned short* ws = (unsigned short*)d_ws;
  unsigned short* wih_pk = ws;                    // 48 tiles x 1 chunk
  unsigned short* whh_pk = ws + 24576;            // 48 x 8
  unsigned short* w0_pk  = ws + 24576 + 196608;   // 8 x 8
  unsigned short* w1_pk  = w0_pk + 32768;         // 8 x 4
  unsigned short* w2_pk  = w1_pk + 16384;         // 256 x 4

  launch_repack(wih, wih_pk,  48, 1,  24, stream);
  launch_repack(whh, whh_pk,  48, 8, 256, stream);
  launch_repack(w0,  w0_pk,    8, 8, 256, stream);
  launch_repack(w1,  w1_pk,    8, 4, 128, stream);
  launch_repack(w2,  w2_pk,  256, 4, 128, stream);

  grucde_main<<<B_ / BQ, NTH, 0, stream>>>(
      y_past, t_arr, cxp, gb, gbn, b0, b1, b2, row_, rob,
      wih_pk, whh_pk, w0_pk, w1_pk, w2_pk, out);
}
